// Attention_66400194396821
// MI455X (gfx1250) — matrix operand order repaired and checked by reading, not compiled, not run on hardware
//
#include <hip/hip_runtime.h>
#include <hip/hip_bf16.h>

typedef _Float16 f16;
typedef __attribute__((ext_vector_type(16))) _Float16 v16h;
typedef __attribute__((ext_vector_type(8)))  _Float16 v8h;
typedef __attribute__((ext_vector_type(8)))  float    v8f;

#define DIM    1024
#define HEADS  16
#define DH     64
#define NSEQ   2048
#define BATCH  2
#define ROWS   (BATCH * NSEQ)      // 4096
#define QKVC   (3 * DIM)           // 3072

__device__ __forceinline__ v16h cat16(v8h a, v8h b) {
    return __builtin_shufflevector(a, b, 0,1,2,3,4,5,6,7,8,9,10,11,12,13,14,15);
}

// ---------------------------------------------------------------------------
// Kernel 1: LayerNorm + cast to f16.  One block (256 thr) per row of 1024.
// ---------------------------------------------------------------------------
__global__ void ln_cast_kernel(const float* __restrict__ x,
                               const float* __restrict__ lw,
                               const float* __restrict__ lb,
                               f16* __restrict__ xn) {
    const int row = blockIdx.x;
    const int t   = threadIdx.x;
    const float* xr = x + (size_t)row * DIM;
    float vals[4], s = 0.f, ss = 0.f;
#pragma unroll
    for (int i = 0; i < 4; ++i) {
        float v = xr[t + i * 256];
        vals[i] = v; s += v; ss += v * v;
    }
    __shared__ float red0[256], red1[256];
    red0[t] = s; red1[t] = ss;
    __syncthreads();
    for (int off = 128; off > 0; off >>= 1) {
        if (t < off) { red0[t] += red0[t + off]; red1[t] += red1[t + off]; }
        __syncthreads();
    }
    const float mean = red0[0] * (1.f / DIM);
    const float var  = red1[0] * (1.f / DIM) - mean * mean;
    const float inv  = rsqrtf(var + 1e-5f);
#pragma unroll
    for (int i = 0; i < 4; ++i) {
        int c = t + i * 256;
        xn[(size_t)row * DIM + c] = (f16)((vals[i] - mean) * inv * lw[c] + lb[c]);
    }
}

// ---------------------------------------------------------------------------
// Kernel 2: cast + transpose weights: W[K][Nc] f32 -> WT[Nc][K] f16
// ---------------------------------------------------------------------------
__global__ void cast_transpose_kernel(const float* __restrict__ W,
                                      f16* __restrict__ WT, int K, int Nc) {
    int idx = blockIdx.x * blockDim.x + threadIdx.x;
    if (idx >= K * Nc) return;
    int k = idx / Nc, n = idx % Nc;
    WT[(size_t)n * K + k] = (f16)W[idx];
}

// ---------------------------------------------------------------------------
// Kernel 3/7: WMMA GEMM.  C[M x Ncols] = A[M x K] * B, B given transposed as
// BT[Ncols x K].  One wave -> 64x64 tile (4x4 frags, 16 WMMA / k-step).
// Block 128 thr = 4 waves in 2x2 -> 128x128 tile.  grid = (M/128, Ncols/128).
// ---------------------------------------------------------------------------
template <bool F32OUT>
__global__ void gemm_wmma_kernel(const f16* __restrict__ A,
                                 const f16* __restrict__ BT,
                                 void* __restrict__ Cout, int K, int Ncols) {
    const int lane = threadIdx.x & 31;
    const int wave = threadIdx.x >> 5;
    const int wr = wave >> 1;
    const int wc = wave & 1;
    const int m  = lane & 15;
    const int hi = lane >> 4;
    const int row0 = blockIdx.x * 128 + wr * 64;
    const int col0 = blockIdx.y * 128 + wc * 64;

    const f16* arow[4];
    const f16* bcol[4];
#pragma unroll
    for (int i = 0; i < 4; ++i) {
        arow[i] = A  + (size_t)(row0 + i * 16 + m) * K + hi * 8;
        bcol[i] = BT + (size_t)(col0 + i * 16 + m) * K + hi * 8;
    }

    v8f acc[4][4];
#pragma unroll
    for (int r = 0; r < 4; ++r)
#pragma unroll
        for (int c = 0; c < 4; ++c) acc[r][c] = v8f{};

    for (int k0 = 0; k0 < K; k0 += 32) {
        if (k0 + 32 < K) {
#pragma unroll
            for (int i = 0; i < 4; ++i) {
                __builtin_prefetch(arow[i] + k0 + 32, 0, 0);
                __builtin_prefetch(bcol[i] + k0 + 32, 0, 0);
            }
        }
        v16h af[4], bf[4];
#pragma unroll
        for (int i = 0; i < 4; ++i) {
            af[i] = cat16(*(const v8h*)(arow[i] + k0),
                          *(const v8h*)(arow[i] + k0 + 16));
            bf[i] = cat16(*(const v8h*)(bcol[i] + k0),
                          *(const v8h*)(bcol[i] + k0 + 16));
        }
#pragma unroll
        for (int r = 0; r < 4; ++r)
#pragma unroll
            for (int c = 0; c < 4; ++c)
                acc[r][c] = __builtin_amdgcn_wmma_f32_16x16x32_f16(
                    false, af[r], false, bf[c], (short)0, acc[r][c], false, false);
    }
#pragma unroll
    for (int r = 0; r < 4; ++r)
#pragma unroll
        for (int c = 0; c < 4; ++c)
#pragma unroll
            for (int e = 0; e < 8; ++e) {
                int rr = row0 + r * 16 + e + 8 * hi;
                int cc = col0 + c * 16 + m;
                if (F32OUT)
                    ((float*)Cout)[(size_t)rr * Ncols + cc] = acc[r][c][e];
                else
                    ((f16*)Cout)[(size_t)rr * Ncols + cc] = (f16)acc[r][c][e];
            }
}

// ---------------------------------------------------------------------------
// Kernel 4: l2-normalize q,k head rows in place; emit V^T [b,h,d,n] f16.
// ---------------------------------------------------------------------------
__global__ void norm_qk_vt_kernel(f16* __restrict__ qkv, f16* __restrict__ vt) {
    const int idx  = blockIdx.x;
    const int bn   = idx >> 4;
    const int h    = idx & 15;
    const int lane = threadIdx.x;
    f16* base = qkv + (size_t)bn * QKVC + h * DH;

#pragma unroll
    for (int part = 0; part < 2; ++part) {
        f16* p = base + part * DIM;
        float v0 = (float)p[lane], v1 = (float)p[lane + 32];
        float ss = v0 * v0 + v1 * v1;
#pragma unroll
        for (int mk = 16; mk >= 1; mk >>= 1) ss += __shfl_xor(ss, mk, 32);
        float inv = 1.f / fmaxf(sqrtf(ss), 1e-12f);
        p[lane]      = (f16)(v0 * inv);
        p[lane + 32] = (f16)(v1 * inv);
    }
    const f16* vb = base + 2 * DIM;
    const int b = bn >> 11, n = bn & (NSEQ - 1);
    f16* vtb = vt + ((size_t)(b * HEADS + h) * DH) * NSEQ + n;
    vtb[(size_t)lane * NSEQ]        = vb[lane];
    vtb[(size_t)(lane + 32) * NSEQ] = vb[lane + 32];
}

// ---------------------------------------------------------------------------
// Flash-attention inner step over one 32-key block.  MASKED instantiation
// only for causal-boundary blocks.  Butterfly reductions are stage-parallel
// across the 8 rows so the ds_bpermutes clause together (1 wait per stage).
// ---------------------------------------------------------------------------
template <bool MASKED>
__device__ __forceinline__ void flash_step(const v16h qf[2],
                                           const f16* __restrict__ kbase,
                                           const f16* __restrict__ vtb,
                                           f16* __restrict__ Pw,
                                           int j0, int i0, int m, int hi,
                                           float mi[8], float li[8], v8f acc[4]) {
    // ---- S = (8*Q) * K^T : two 16x16 tiles over 32 keys ----
    v8f st[2] = {v8f{}, v8f{}};
#pragma unroll
    for (int t = 0; t < 2; ++t) {
        const f16* krow = kbase + (size_t)(j0 + t * 16 + m) * QKVC;
#pragma unroll
        for (int c = 0; c < 2; ++c) {
            const f16* p = krow + c * 32 + hi * 8;
            v16h bf = cat16(*(const v8h*)p, *(const v8h*)(p + 16));
            st[t] = __builtin_amdgcn_wmma_f32_16x16x32_f16(
                false, qf[c], false, bf, (short)0, st[t], false, false);
        }
    }
    // ---- mask + row max (stage-parallel butterfly over 16-lane group) ----
    float sv0[8], sv1[8], vmax[8];
#pragma unroll
    for (int r = 0; r < 8; ++r) {
        float a = st[0][r], b = st[1][r];
        if (MASKED) {
            const int irow = i0 + r + 8 * hi;
            if (j0 + m > irow)      a = -1e30f;
            if (j0 + 16 + m > irow) b = -1e30f;
        }
        sv0[r] = a; sv1[r] = b;
        vmax[r] = fmaxf(a, b);
    }
#pragma unroll
    for (int mk = 8; mk >= 1; mk >>= 1) {
        float tmp[8];
#pragma unroll
        for (int r = 0; r < 8; ++r) tmp[r] = __shfl_xor(vmax[r], mk, 32);
#pragma unroll
        for (int r = 0; r < 8; ++r) vmax[r] = fmaxf(vmax[r], tmp[r]);
    }
    // ---- exp + row sum (stage-parallel butterfly) ----
    float nm[8], p0[8], p1[8], rs[8];
#pragma unroll
    for (int r = 0; r < 8; ++r) {
        nm[r] = fmaxf(mi[r], vmax[r]);
        p0[r] = __expf(sv0[r] - nm[r]);
        p1[r] = __expf(sv1[r] - nm[r]);
        rs[r] = p0[r] + p1[r];
    }
#pragma unroll
    for (int mk = 8; mk >= 1; mk >>= 1) {
        float tmp[8];
#pragma unroll
        for (int r = 0; r < 8; ++r) tmp[r] = __shfl_xor(rs[r], mk, 32);
#pragma unroll
        for (int r = 0; r < 8; ++r) rs[r] += tmp[r];
    }
    // ---- online update ----
#pragma unroll
    for (int r = 0; r < 8; ++r) {
        const float sf = __expf(mi[r] - nm[r]);
        li[r] = li[r] * sf + rs[r];
        mi[r] = nm[r];
#pragma unroll
        for (int s = 0; s < 4; ++s) acc[s][r] *= sf;
    }
    // ---- transpose P via LDS: C-layout -> A-layout (16x32) ----
#pragma unroll
    for (int r = 0; r < 8; ++r) {
        Pw[(r + 8 * hi) * 32 + m]      = (f16)p0[r];
        Pw[(r + 8 * hi) * 32 + 16 + m] = (f16)p1[r];
    }
    asm volatile("s_wait_dscnt 0x0" ::: "memory");
    const f16* pr = &Pw[m * 32 + hi * 8];
    v16h pf = cat16(*(const v8h*)pr, *(const v8h*)(pr + 16));
    // ---- O += P * V (V read K-contiguous from V^T) ----
#pragma unroll
    for (int s = 0; s < 4; ++s) {
        const f16* vp = vtb + (size_t)(s * 16 + m) * NSEQ + j0 + hi * 8;
        v16h vf = cat16(*(const v8h*)vp, *(const v8h*)(vp + 16));
        acc[s] = __builtin_amdgcn_wmma_f32_16x16x32_f16(
            false, pf, false, vf, (short)0, acc[s], false, false);
    }
    asm volatile("s_wait_dscnt 0x0" ::: "memory");
}

// ---------------------------------------------------------------------------
// Kernel 5: causal flash attention.  4 waves / block; wave = one
// (b, h, 16-query tile).  Unmasked blocks take the mask-free path.
// ---------------------------------------------------------------------------
__global__ void attn_kernel(const f16* __restrict__ qkv,
                            const f16* __restrict__ vt,
                            f16* __restrict__ obuf) {
    __shared__ __align__(16) f16 Pl[4][16 * 32];

    const int wave = threadIdx.x >> 5;
    const int blk  = blockIdx.x * 4 + wave;
    const int it   = blk & 127;
    const int h    = (blk >> 7) & 15;
    const int b    = blk >> 11;
    const int lane = threadIdx.x & 31;
    const int m  = lane & 15;
    const int hi = lane >> 4;
    const int i0 = it * 16;
    f16* Pw = Pl[wave];

    const f16* qrow = qkv + ((size_t)(b * NSEQ) + i0 + m) * QKVC + h * DH;
    v16h qf[2];
#pragma unroll
    for (int c = 0; c < 2; ++c) {
        const f16* p = qrow + c * 32 + hi * 8;
        v16h q = cat16(*(const v8h*)p, *(const v8h*)(p + 16));
        qf[c] = q * (_Float16)8.0f;      // fold attention scale (exact, 2^3)
    }

    float mi[8], li[8];
    v8f acc[4] = {v8f{}, v8f{}, v8f{}, v8f{}};
#pragma unroll
    for (int r = 0; r < 8; ++r) { mi[r] = -1e30f; li[r] = 0.f; }

    const f16* kbase = qkv + (size_t)(b * NSEQ) * QKVC + DIM + h * DH;
    const f16* vtb   = vt + (size_t)(b * HEADS + h) * DH * NSEQ;
    const int nfull   = i0 >> 5;              // fully-unmasked 32-key blocks
    const int jb_last = (i0 + 15) >> 5;

    for (int jb = 0; jb < nfull; ++jb)
        flash_step<false>(qf, kbase, vtb, Pw, jb * 32, i0, m, hi, mi, li, acc);
    for (int jb = nfull; jb <= jb_last; ++jb)
        flash_step<true >(qf, kbase, vtb, Pw, jb * 32, i0, m, hi, mi, li, acc);

    // ---- epilogue: divide by row sums, store O as f16 [row, h*64 + d] ----
#pragma unroll
    for (int s = 0; s < 4; ++s)
#pragma unroll
        for (int r = 0; r < 8; ++r) {
            const int irow = i0 + r + 8 * hi;
            float o = acc[s][r] / fmaxf(li[r], 1e-20f);
            obuf[((size_t)(b * NSEQ) + irow) * DIM + h * DH + s * 16 + m] = (f16)o;
        }
}

// ---------------------------------------------------------------------------
extern "C" void kernel_launch(void* const* d_in, const int* in_sizes, int n_in,
                              void* d_out, int out_size, void* d_ws, size_t ws_size,
                              hipStream_t stream) {
    const float* x    = (const float*)d_in[0];
    const float* ln_w = (const float*)d_in[1];
    const float* ln_b = (const float*)d_in[2];
    const float* Wqkv = (const float*)d_in[3];
    const float* Wout = (const float*)d_in[4];
    float* out = (float*)d_out;

    char* ws = (char*)d_ws;
    f16* xn    = (f16*)(ws + 0);                         //  8.0 MB  [4096,1024]
    f16* wqkvT = (f16*)(ws + 8388608ull);                //  6.0 MB  [3072,1024]
    f16* woutT = (f16*)(ws + 14680064ull);               //  2.0 MB  [1024,1024]
    f16* qkv   = (f16*)(ws + 16777216ull);               // 24.0 MB  [4096,3072]
    f16* vt    = (f16*)(ws + 41943040ull);               //  8.0 MB  [b,h,d,n]
    f16* obuf  = (f16*)(ws + 50331648ull);               //  8.0 MB  [4096,1024]

    ln_cast_kernel<<<ROWS, 256, 0, stream>>>(x, ln_w, ln_b, xn);
    cast_transpose_kernel<<<(DIM * QKVC) / 256, 256, 0, stream>>>(Wqkv, wqkvT, DIM, QKVC);
    cast_transpose_kernel<<<(DIM * DIM) / 256, 256, 0, stream>>>(Wout, woutT, DIM, DIM);
    gemm_wmma_kernel<false><<<dim3(ROWS / 128, QKVC / 128), 128, 0, stream>>>(
        xn, wqkvT, (void*)qkv, DIM, QKVC);
    norm_qk_vt_kernel<<<ROWS * HEADS, 32, 0, stream>>>(qkv, vt);
    attn_kernel<<<BATCH * HEADS * (NSEQ / 16) / 4, 128, 0, stream>>>(qkv, vt, obuf);
    gemm_wmma_kernel<true><<<dim3(ROWS / 128, DIM / 128), 128, 0, stream>>>(
        obuf, woutT, (void*)out, DIM, DIM);
}
